// NerfHead_73392401154483
// MI455X (gfx1250) — matrix operand order repaired and checked by reading, not compiled, not run on hardware
//
#include <hip/hip_runtime.h>
#include <hip/hip_bf16.h>
#include <stdint.h>

// ---------------------------------------------------------------------------
// MI455X (gfx1250) implementation.
//
//  pixels  [2048,256,128] f32   patches [2048,768] f32
//  W_pg    [32768,768]    f32   b_pg    [32768]    f32   norm_w [128] f32
//
//  Kernel 1: params[2048,32768] = patches @ W_pg^T + b   (bf16 WMMA, f32 acc,
//            double-buffered LDS + register prefetch)
//  Kernel 2: per-sample fused: L2-colnorm(layer1), RMSNorm(pixels),
//            h = silu(x@L1), out = h@L2 + pixels          (bf16 WMMA, f32 acc)
// ---------------------------------------------------------------------------

typedef __attribute__((ext_vector_type(16))) __bf16 v16bf;
typedef __attribute__((ext_vector_type(2)))  __bf16 v2bf;
typedef __attribute__((ext_vector_type(8)))  float  v8f;

union FragB {            // one WMMA 16-bit operand: 32 bytes / lane
    v16bf v;
    uint4 q[2];
};

// Native f32->bf16 (backend selects v_cvt_pk_bf16_f32-class ops on gfx1250)
__device__ __forceinline__ uint32_t pack2(float a, float b) {
    v2bf p;
    p.x = (__bf16)a;
    p.y = (__bf16)b;
    return __builtin_bit_cast(uint32_t, p);
}
__device__ __forceinline__ unsigned short f2bf(float f) {
    __bf16 h = (__bf16)f;
    return __builtin_bit_cast(unsigned short, h);
}
__device__ __forceinline__ uint2 pack4(float4 v) {
    return make_uint2(pack2(v.x, v.y), pack2(v.z, v.w));
}

// ---------------------------------------------------------------------------
// Kernel 1: params = patches @ W_pg^T + b_pg
// Block: 256 threads (8 waves), tile 64M x 128N, K-step 32, 24 K-steps.
// Wave w owns N-strip w*16 and all 4 M sub-tiles (4 accumulators).
// Double-buffered LDS; global prefetch of K-step k+1 overlaps the WMMAs of k.
// ---------------------------------------------------------------------------
__global__ __launch_bounds__(256)
void pg_gemm_kernel(const float* __restrict__ patches,
                    const float* __restrict__ W,
                    const float* __restrict__ bias,
                    float* __restrict__ params)
{
    __shared__ alignas(16) unsigned short As[2][64 * 32];   // [m][k] bf16
    __shared__ alignas(16) unsigned short Bs[2][128 * 32];  // [n][k] bf16

    const int t  = threadIdx.x;
    const int l  = t & 31;
    const int w  = t >> 5;
    const int ml = l & 15;
    const int g  = l >> 4;
    const int m0 = blockIdx.y * 64;
    const int n0 = blockIdx.x * 128;

    // per-thread staging assignment
    const int ar = (t * 8) >> 5, ac = (t * 8) & 31;    // A: 8 floats
    const int bn = t >> 1,       bc = (t & 1) * 16;    // B: 16 floats
    const float* aptr = patches + (size_t)(m0 + ar) * 768 + ac;
    const float* bptr = W       + (size_t)(n0 + bn) * 768 + bc;

    float4 pa0, pa1, pb0, pb1, pb2, pb3;

    // prologue: fetch K-step 0 and stage into buffer 0
    pa0 = ((const float4*)aptr)[0];
    pa1 = ((const float4*)aptr)[1];
    pb0 = ((const float4*)bptr)[0];
    pb1 = ((const float4*)bptr)[1];
    pb2 = ((const float4*)bptr)[2];
    pb3 = ((const float4*)bptr)[3];
    {
        uint2* da = (uint2*)(As[0] + ar * 32 + ac);
        da[0] = pack4(pa0); da[1] = pack4(pa1);
        uint2* db = (uint2*)(Bs[0] + bn * 32 + bc);
        db[0] = pack4(pb0); db[1] = pack4(pb1);
        db[2] = pack4(pb2); db[3] = pack4(pb3);
    }
    __syncthreads();

    v8f acc[4] = {};

    for (int kt = 0; kt < 24; ++kt) {
        const int cur = kt & 1, nxt = cur ^ 1;

        // issue global prefetch of next K-step first (hidden behind WMMAs)
        if (kt < 23) {
            const int k1 = (kt + 1) * 32;
            pa0 = ((const float4*)(aptr + k1))[0];
            pa1 = ((const float4*)(aptr + k1))[1];
            pb0 = ((const float4*)(bptr + k1))[0];
            pb1 = ((const float4*)(bptr + k1))[1];
            pb2 = ((const float4*)(bptr + k1))[2];
            pb3 = ((const float4*)(bptr + k1))[3];
        }

        // B fragment (ISA layout: lane N = l&15, K = 8*(l>>4)+s, 16+8*(l>>4)+s)
        FragB bf;
        const unsigned short* brow = Bs[cur] + (w * 16 + ml) * 32;
        bf.q[0] = *(const uint4*)(brow + g * 8);
        bf.q[1] = *(const uint4*)(brow + 16 + g * 8);

        #pragma unroll
        for (int mt = 0; mt < 4; ++mt) {
            FragB af;
            const unsigned short* arow = As[cur] + (mt * 16 + ml) * 32;
            af.q[0] = *(const uint4*)(arow + g * 8);
            af.q[1] = *(const uint4*)(arow + 16 + g * 8);
            acc[mt] = __builtin_amdgcn_wmma_f32_16x16x32_bf16(
                false, af.v, false, bf.v, (short)0, acc[mt], false, false);
        }

        // stage prefetched tile into the other buffer
        if (kt < 23) {
            uint2* da = (uint2*)(As[nxt] + ar * 32 + ac);
            da[0] = pack4(pa0); da[1] = pack4(pa1);
            uint2* db = (uint2*)(Bs[nxt] + bn * 32 + bc);
            db[0] = pack4(pb0); db[1] = pack4(pb1);
            db[2] = pack4(pb2); db[3] = pack4(pb3);
        }
        __syncthreads();
    }

    const int N = n0 + w * 16 + ml;
    const float bv = bias[N];
    #pragma unroll
    for (int mt = 0; mt < 4; ++mt) {
        #pragma unroll
        for (int r = 0; r < 8; ++r) {
            const int M = m0 + mt * 16 + r + 8 * g;   // C layout: M = r + 8*(l>>4)
            params[(size_t)M * 32768 + N] = acc[mt][r] + bv;
        }
    }
}

// ---------------------------------------------------------------------------
// Kernel 2: per-sample fused normalize + 2-layer MLP with SiLU + residual.
// One workgroup (256 threads, 8 waves) per sample.  LDS stride 136 ushorts
// (272B) keeps ds_load_b128 16B-aligned and spreads banks.
// ---------------------------------------------------------------------------
#define LSTR 136

__global__ __launch_bounds__(256)
void fused_mlp_kernel(const float* __restrict__ pixels,
                      const float* __restrict__ params,
                      const float* __restrict__ norm_w,
                      float* __restrict__ out)
{
    __shared__ alignas(16) unsigned short L1[128 * LSTR];  // [e][d] bf16 (normalized, transposed)
    __shared__ alignas(16) unsigned short L2b[128 * LSTR]; // [f][e] bf16 (transposed)
    __shared__ alignas(16) unsigned short Xb[64 * LSTR];   // [p][d] bf16 (RMS-normed pixels)
    __shared__ alignas(16) unsigned short Hb[64 * LSTR];   // [p][e] bf16 (silu(h))
    __shared__ float red[256];
    __shared__ float scl[128];

    const int t  = threadIdx.x;
    const int l  = t & 31;
    const int w  = t >> 5;
    const int ml = l & 15;
    const int g  = l >> 4;
    const int b  = blockIdx.x;

    const float* p1 = params + (size_t)b * 32768;          // layer1 [d][e]
    const float* p2 = p1 + 16384;                          // layer2 [e][f]

    // ---- column L2 norms of layer1 (norm over rows d, per column e) ----
    {
        const int j = t & 127, half = t >> 7;
        float s = 0.f;
        for (int i = half * 64; i < half * 64 + 64; ++i) {
            float v = p1[i * 128 + j];
            s += v * v;
        }
        red[t] = s;
        __syncthreads();
        if (t < 128) {
            float nrm = sqrtf(red[t] + red[t + 128]);
            scl[t] = 1.0f / fmaxf(nrm, 1e-12f);
        }
        __syncthreads();
    }
    // ---- stage layer1 (normalized) and layer2, transposed to [n][k] bf16 ----
    {
        const int j = t & 127, half = t >> 7;
        const float sj = scl[j];
        for (int i = half * 64; i < half * 64 + 64; ++i) {
            L1[j * LSTR + i]  = f2bf(p1[i * 128 + j] * sj);
            L2b[j * LSTR + i] = f2bf(p2[i * 128 + j]);
        }
    }
    __syncthreads();

    const int mt = w & 3;            // wave's 16-row M sub-tile in the chunk
    const int nb = (w >> 2) * 4;     // wave's 4 N sub-tiles

    for (int c0 = 0; c0 < 256; c0 += 64) {
        // ---- RMSNorm 64 pixel rows: one row per wave pass, wave32 reduce ----
        #pragma unroll
        for (int rr = 0; rr < 8; ++rr) {
            const int p = w * 8 + rr;
            const float4 v = ((const float4*)(pixels +
                              ((size_t)b * 256 + c0 + p) * 128))[l];
            float ss = v.x * v.x + v.y * v.y + v.z * v.z + v.w * v.w;
            #pragma unroll
            for (int o = 16; o; o >>= 1) ss += __shfl_xor(ss, o, 32);
            const float sc = rsqrtf(ss * (1.0f / 128.0f) + 1.1920928955078125e-07f);
            const float4 nw = ((const float4*)norm_w)[l];
            *(uint2*)(Xb + p * LSTR + l * 4) =
                make_uint2(pack2(v.x * sc * nw.x, v.y * sc * nw.y),
                           pack2(v.z * sc * nw.z, v.w * sc * nw.w));
        }
        __syncthreads();

        // ---- GEMM1: h = x @ L1, then SiLU -> Hb ----
        v8f acc[4] = {};
        #pragma unroll
        for (int kt = 0; kt < 4; ++kt) {
            FragB af;
            const unsigned short* arow = Xb + (mt * 16 + ml) * LSTR + kt * 32;
            af.q[0] = *(const uint4*)(arow + g * 8);
            af.q[1] = *(const uint4*)(arow + 16 + g * 8);
            #pragma unroll
            for (int jn = 0; jn < 4; ++jn) {
                FragB bf;
                const unsigned short* brow =
                    L1 + ((nb + jn) * 16 + ml) * LSTR + kt * 32;
                bf.q[0] = *(const uint4*)(brow + g * 8);
                bf.q[1] = *(const uint4*)(brow + 16 + g * 8);
                acc[jn] = __builtin_amdgcn_wmma_f32_16x16x32_bf16(
                    false, af.v, false, bf.v, (short)0, acc[jn], false, false);
            }
        }
        #pragma unroll
        for (int jn = 0; jn < 4; ++jn) {
            const int N = (nb + jn) * 16 + ml;
            #pragma unroll
            for (int r = 0; r < 8; ++r) {
                const int M = mt * 16 + r + 8 * g;
                const float h = acc[jn][r];
                Hb[M * LSTR + N] = f2bf(h / (1.0f + __expf(-h)));
            }
        }
        __syncthreads();

        // ---- GEMM2: out = silu(h) @ L2 + pixels ----
        v8f acc2[4] = {};
        #pragma unroll
        for (int kt = 0; kt < 4; ++kt) {
            FragB af;
            const unsigned short* arow = Hb + (mt * 16 + ml) * LSTR + kt * 32;
            af.q[0] = *(const uint4*)(arow + g * 8);
            af.q[1] = *(const uint4*)(arow + 16 + g * 8);
            #pragma unroll
            for (int jn = 0; jn < 4; ++jn) {
                FragB bf;
                const unsigned short* brow =
                    L2b + ((nb + jn) * 16 + ml) * LSTR + kt * 32;
                bf.q[0] = *(const uint4*)(brow + g * 8);
                bf.q[1] = *(const uint4*)(brow + 16 + g * 8);
                acc2[jn] = __builtin_amdgcn_wmma_f32_16x16x32_bf16(
                    false, af.v, false, bf.v, (short)0, acc2[jn], false, false);
            }
        }
        #pragma unroll
        for (int jn = 0; jn < 4; ++jn) {
            const int N = (nb + jn) * 16 + ml;
            #pragma unroll
            for (int r = 0; r < 8; ++r) {
                const int M = mt * 16 + r + 8 * g;
                const size_t idx = ((size_t)b * 256 + c0 + M) * 128 + N;
                out[idx] = acc2[jn][r] + pixels[idx];
            }
        }
        __syncthreads();   // Xb/Hb reused next chunk
    }
}

// ---------------------------------------------------------------------------
extern "C" void kernel_launch(void* const* d_in, const int* in_sizes, int n_in,
                              void* d_out, int out_size, void* d_ws, size_t ws_size,
                              hipStream_t stream)
{
    const float* pixels  = (const float*)d_in[0];
    const float* patches = (const float*)d_in[1];
    const float* W_pg    = (const float*)d_in[2];
    const float* b_pg    = (const float*)d_in[3];
    const float* norm_w  = (const float*)d_in[4];
    float*       out     = (float*)d_out;
    float*       params  = (float*)d_ws;   // 2048*32768 f32 = 268 MB scratch

    // params = patches @ W_pg^T + b : grid = (32768/128, 2048/64)
    pg_gemm_kernel<<<dim3(256, 32), 256, 0, stream>>>(patches, W_pg, b_pg, params);

    // fused per-sample MLP: one block per sample
    fused_mlp_kernel<<<dim3(2048), 256, 0, stream>>>(pixels, params, norm_w, out);
}
